// PAM_77318001262618
// MI455X (gfx1250) — matrix operand order repaired and checked by reading, not compiled, not run on hardware
//
#include <hip/hip_runtime.h>

typedef __bf16 bf16_t;
typedef __attribute__((ext_vector_type(16))) __bf16 v16bf;
typedef __attribute__((ext_vector_type(8)))  __bf16 v8bf;
typedef __attribute__((ext_vector_type(8)))  float  v8f;
typedef __attribute__((ext_vector_type(4)))  float  v4f;
typedef __attribute__((ext_vector_type(4)))  unsigned u32x4;
typedef __attribute__((ext_vector_type(4)))  int      i32x4;
typedef __attribute__((ext_vector_type(8)))  int      i32x8;

#define NB   8
#define NN   4096
#define CC   512
#define DD   64
#define NW   640   // 64 (Q) + 64 (K) + 512 (V) packed projection columns
#define QT   32    // q rows per workgroup (attention)
#define KT   128   // key tile (attention)

#if __has_builtin(__builtin_amdgcn_global_load_async_to_lds_b128)
#define HAVE_ASYNC 1
#endif
#if __has_builtin(__builtin_amdgcn_tensor_load_to_lds)
#define HAVE_TDM 1
#endif

// ---------------- helpers ----------------
__device__ __forceinline__ unsigned short bfbits(float f) {
  unsigned u = __builtin_bit_cast(unsigned, f);
  u += 0x7FFFu + ((u >> 16) & 1u);      // round-to-nearest-even
  return (unsigned short)(u >> 16);
}
__device__ __forceinline__ bf16_t f2bf(float f) {
  unsigned short s = bfbits(f);
  return __builtin_bit_cast(bf16_t, s);
}
__device__ __forceinline__ v16bf cat8(v8bf lo, v8bf hi) {
  return __builtin_shufflevector(lo, hi, 0,1,2,3,4,5,6,7,8,9,10,11,12,13,14,15);
}

// 16B global -> LDS copy: async DMA on CDNA5, plain copy fallback
__device__ __forceinline__ void cp16_g2l(const bf16_t* g, bf16_t* l) {
#ifdef HAVE_ASYNC
  __builtin_amdgcn_global_load_async_to_lds_b128(
      (__attribute__((address_space(1))) i32x4*)(g),
      (__attribute__((address_space(3))) i32x4*)(l), 0, 0);
#else
  *(v8bf*)l = *(const v8bf*)g;
#endif
}
__device__ __forceinline__ void cp_wait() {
#ifdef HAVE_ASYNC
#if __has_builtin(__builtin_amdgcn_s_wait_asynccnt)
  __builtin_amdgcn_s_wait_asynccnt(0);
#else
  asm volatile("s_wait_asynccnt 0x0" ::: "memory");
#endif
#endif
}

// ---------------- kernel 0a: fuse + transpose weights to bf16 ----------------
// Wt[j][k] (j = packed output col 0..639, k = input channel 0..511)
__global__ void pack_weights(const float* __restrict__ Wb, const float* __restrict__ Wc,
                             const float* __restrict__ Wd, bf16_t* __restrict__ Wt) {
  int idx = blockIdx.x * 256 + threadIdx.x;
  if (idx >= NW * CC) return;
  int j = idx >> 9, k = idx & (CC - 1);
  float v;
  if (j < DD)        v = Wb[k * DD + j];
  else if (j < 2*DD) v = Wc[k * DD + (j - DD)];
  else               v = Wd[k * CC + (j - 2*DD)];
  Wt[idx] = f2bf(v);
}

// ---------------- kernel 0b: x (f32) -> xb (bf16), streaming ----------------
__global__ void xcast(const float* __restrict__ x, bf16_t* __restrict__ xb) {
  long i = ((long)blockIdx.x * 256 + threadIdx.x) * 8;
  v4f a = *(const v4f*)&x[i];
  v4f b = *(const v4f*)&x[i + 4];
  u32x4 p;
  p[0] = (unsigned)bfbits(a[0]) | ((unsigned)bfbits(a[1]) << 16);
  p[1] = (unsigned)bfbits(a[2]) | ((unsigned)bfbits(a[3]) << 16);
  p[2] = (unsigned)bfbits(b[0]) | ((unsigned)bfbits(b[1]) << 16);
  p[3] = (unsigned)bfbits(b[2]) | ((unsigned)bfbits(b[3]) << 16);
  *(u32x4*)&xb[i] = p;
}

// ---------------- kernel 1: QKV projection (bf16 WMMA, f32 accum) ----------------
// grid: (B*N)/32 blocks, 256 threads (8 waves). Each block: 32 rows x 640 cols.
// A tile staged into LDS by the Tensor Data Mover (32x32 bf16 2D tile).
// V is written TRANSPOSED (Vt[b][c][n]) so attention B-fragments are contiguous.
__global__ __launch_bounds__(256) void qkv_proj(
    const bf16_t* __restrict__ xb, const bf16_t* __restrict__ Wt,
    bf16_t* __restrict__ Q, bf16_t* __restrict__ K, bf16_t* __restrict__ Vt) {
  __shared__ bf16_t sA[32 * 32];
  const int t = threadIdx.x;
  const int w = t >> 5, lane = t & 31;
  const int m = lane & 15, hi = lane >> 4;
  const long R = (long)blockIdx.x * 32;

  v8f acc[2][5] = {};

  for (int ks = 0; ks < CC; ks += 32) {
    __syncthreads();
#ifdef HAVE_TDM
    if (w == 0) {   // one wave drives the TDM for the whole workgroup
      unsigned long long ga = (unsigned long long)(xb + R * CC + ks);
      unsigned ldsOff = (unsigned)(unsigned long long)
          (__attribute__((address_space(3))) void*)sA;
      // D# group0: count=1 | lds_addr | global_addr | type=2 ("image")
      u32x4 g0 = { 1u, ldsOff, (unsigned)ga,
                   ((unsigned)(ga >> 32) & 0x01FFFFFFu) | 0x80000000u };
      // D# group1: data_size=2B; tensor 512 x 32768, stride 512; tile 32 x 32
      i32x8 g1 = { (int)(1u << 16),     // [17:16] data_size = 1 (2 bytes)
                   (int)(512u << 16),   // tensor_dim0[15:0] = 512
                   (int)0x80000000u,    // tensor_dim1[15:0] = 32768
                   (int)(32u << 16),    // tile_dim0 = 32
                   (int)32,             // tile_dim1 = 32, tile_dim2 = 0
                   (int)512,            // tensor_dim0_stride = 512
                   0, 0 };
      i32x4 zz4 = { 0, 0, 0, 0 };
      i32x8 zz8 = { 0, 0, 0, 0, 0, 0, 0, 0 };
      __builtin_amdgcn_tensor_load_to_lds(g0, g1, zz4, zz4, zz8, 0);
      __builtin_amdgcn_s_wait_tensorcnt(0);
    }
#else
    {   // cooperative copy fallback (xb is already bf16)
      int row = t >> 3, c4 = (t & 7) * 4;
      *(uint2*)&sA[row * 32 + c4] = *(const uint2*)&xb[(R + row) * CC + ks + c4];
    }
#endif
    __syncthreads();

    // A fragments: lane half hi holds 16 contiguous k (16*hi..16*hi+15), same order as B
    v16bf a[2];
#pragma unroll
    for (int rg = 0; rg < 2; ++rg) {
      v8bf l0 = *(const v8bf*)&sA[(rg*16 + m) * 32 + hi*16];
      v8bf l1 = *(const v8bf*)&sA[(rg*16 + m) * 32 + hi*16 + 8];
      a[rg] = cat8(l0, l1);
    }
#pragma unroll
    for (int cg = 0; cg < 5; ++cg) {
      int j = w * 80 + cg * 16 + m;   // B fragment: lane = out col, 16 contiguous k
      v16bf bfrag = *(const v16bf*)&Wt[(long)j * CC + ks + hi*16];
#pragma unroll
      for (int rg = 0; rg < 2; ++rg)
        acc[rg][cg] = __builtin_amdgcn_wmma_f32_16x16x32_bf16(
            false, a[rg], false, bfrag, (short)0, acc[rg][cg], false, false);
    }
  }

  // epilogue: scatter to Q / K (row-major) and Vt (transposed)
#pragma unroll
  for (int rg = 0; rg < 2; ++rg) {
#pragma unroll
    for (int cg = 0; cg < 5; ++cg) {
      int jb = w * 80 + cg * 16;
      int j  = jb + m;
#pragma unroll
      for (int r = 0; r < 8; ++r) {
        long gr = R + rg*16 + r + 8*hi;
        bf16_t val = f2bf(acc[rg][cg][r]);
        if (jb < 64)       Q[gr * 64 + j]        = val;
        else if (jb < 128) K[gr * 64 + (j - 64)] = val;
        else {
          long bbb = gr >> 12, n = gr & (NN - 1);
          Vt[(bbb * CC + (j - 128)) * NN + n] = val;   // Vt[b][c][n]
        }
      }
    }
  }
}

// ---------------- kernel 2: flash attention + residual ----------------
// grid: (N/QT, B), 256 threads (8 waves). Wave w owns out channels [64w, 64w+64).
// K tiles double-buffered through LDS with async global->LDS DMA.
__global__ __launch_bounds__(256) void attn(
    const bf16_t* __restrict__ Q, const bf16_t* __restrict__ K,
    const bf16_t* __restrict__ Vt, const float* __restrict__ x,
    const float* __restrict__ gamma, float* __restrict__ out) {
  extern __shared__ char smem[];
  float*  sS  = (float*)(smem);               // 32 x 128 scores (f32)   16384 B
  bf16_t* sP  = (bf16_t*)(smem + 16384);      // 32 x 128 probs (bf16)    8192 B
  float*  sM  = (float*)(smem + 24576);       // running max [32]
  float*  sL  = (float*)(smem + 24704);       // running sum [32]
  float*  sAl = (float*)(smem + 24832);       // per-tile alpha [32]
  bf16_t* sK0 = (bf16_t*)(smem + 25088);      // K tile buf 0: 128 x 64  16384 B
  bf16_t* sK1 = (bf16_t*)(smem + 41472);      // K tile buf 1: 128 x 64  16384 B

  const int t = threadIdx.x;
  const int w = t >> 5, lane = t & 31;
  const int m = lane & 15, hi = lane >> 4;
  const int bb = blockIdx.y;
  const int qbase = blockIdx.x * QT;
  const int srg = w >> 2;                     // this wave's score row-group (0/1)
  const int kcp = (w & 3) * 2;                // this wave's score key-chunk pair

  const bf16_t* Qb  = Q  + ((long)bb * NN + qbase) * DD;
  const bf16_t* Kb  = K  + (long)bb * NN * DD;
  const bf16_t* VtB = Vt + (long)bb * CC * NN;

  if (t < QT) { sM[t] = -1e30f; sL[t] = 0.f; }

  // prologue: start async copy of K tile 0 (contiguous 16KB)
#pragma unroll
  for (int i = 0; i < 4; ++i)
    cp16_g2l(Kb + (i * 256 + t) * 8, sK0 + (i * 256 + t) * 8);

  // Q A-fragments for this wave's score rows, kept in VGPRs for the whole kernel
  v16bf qa[2];
#pragma unroll
  for (int d2 = 0; d2 < 2; ++d2) {
    const bf16_t* qrow = Qb + (srg*16 + m) * DD + d2*32 + hi*16;
    qa[d2] = cat8(*(const v8bf*)qrow, *(const v8bf*)(qrow + 8));
  }

  v8f acc[2][4] = {};

  for (int it = 0; it < NN / KT; ++it) {
    const int keybase = it * KT;
    bf16_t* sKr = (it & 1) ? sK1 : sK0;
    bf16_t* sKw = (it & 1) ? sK0 : sK1;

    cp_wait();             // current K tile landed in LDS (this wave's copies)
    __syncthreads();       // ... and everyone else's; also publishes sM/sL init

    // kick off next tile's K copy; overlaps with all compute below
    if (it + 1 < NN / KT) {
      const bf16_t* gsrc = Kb + (long)(keybase + KT) * DD;
#pragma unroll
      for (int i = 0; i < 4; ++i)
        cp16_g2l(gsrc + (i * 256 + t) * 8, sKw + (i * 256 + t) * 8);
    }

    // ---- S = Q K^T for this tile: each wave computes 2 of 16 fragments ----
#pragma unroll
    for (int kcs = 0; kcs < 2; ++kcs) {
      int kc = kcp + kcs;
      v8f s = {};
#pragma unroll
      for (int d2 = 0; d2 < 2; ++d2) {
        // B fragment = K^T: lane = key, 16 contiguous d's from the LDS tile
        const bf16_t* krow = sKr + (kc*16 + m) * DD + d2*32 + hi*16;
        v16bf kf = *(const v16bf*)krow;
        s = __builtin_amdgcn_wmma_f32_16x16x32_bf16(false, qa[d2], false, kf,
                                                    (short)0, s, false, false);
      }
#pragma unroll
      for (int r = 0; r < 8; ++r)
        sS[(srg*16 + r + 8*hi) * KT + kc*16 + m] = s[r];
    }
    __syncthreads();

    // ---- online softmax over the tile (8 threads per row) ----
    {
      int row = t >> 3, j = t & 7;
      float* srow = &sS[row * KT + j*16];
      float mx = srow[0];
#pragma unroll
      for (int i = 1; i < 16; ++i) mx = fmaxf(mx, srow[i]);
#pragma unroll
      for (int off = 1; off < 8; off <<= 1) mx = fmaxf(mx, __shfl_xor(mx, off, 8));
      float mo = sM[row];
      float mn = fmaxf(mo, mx);
      float al = __expf(mo - mn);
      float sum = 0.f;
      bf16_t* prow = &sP[row * KT + j*16];
#pragma unroll
      for (int i = 0; i < 16; ++i) {
        float e = __expf(srow[i] - mn);
        sum += e;
        prow[i] = f2bf(e);
      }
#pragma unroll
      for (int off = 1; off < 8; off <<= 1) sum += __shfl_xor(sum, off, 8);
      if (j == 0) { sM[row] = mn; sL[row] = sL[row] * al + sum; sAl[row] = al; }
    }
    __syncthreads();

    // ---- rescale accumulators, then acc += P @ V_tile (V^T from global) ----
#pragma unroll
    for (int rg = 0; rg < 2; ++rg) {
      v8f al;
#pragma unroll
      for (int r = 0; r < 8; ++r) al[r] = sAl[rg*16 + r + 8*hi];
#pragma unroll
      for (int cg = 0; cg < 4; ++cg) acc[rg][cg] *= al;
    }
#pragma unroll
    for (int ks = 0; ks < 4; ++ks) {
      v16bf pa[2];
#pragma unroll
      for (int rg = 0; rg < 2; ++rg) {
        const bf16_t* pr = &sP[(rg*16 + m) * KT + ks*32 + hi*16];
        pa[rg] = cat8(*(const v8bf*)pr, *(const v8bf*)(pr + 8));
      }
#pragma unroll
      for (int cg = 0; cg < 4; ++cg) {
        // B fragment: lane = out channel, 16 contiguous keys (Vt row-major)
        const bf16_t* vr = VtB + (long)(w*64 + cg*16 + m) * NN
                               + keybase + ks*32 + hi*16;
        if (ks == 0 && keybase + KT < NN) __builtin_prefetch(vr + KT, 0, 0);
        v16bf vf = *(const v16bf*)vr;
#pragma unroll
        for (int rg = 0; rg < 2; ++rg)
          acc[rg][cg] = __builtin_amdgcn_wmma_f32_16x16x32_bf16(
              false, pa[rg], false, vf, (short)0, acc[rg][cg], false, false);
      }
    }
    // no end-of-iteration barrier needed: next writes to sS/sP/sK are each
    // separated from this iteration's reads by >=1 barrier above
  }

  // ---- epilogue: out = gamma * (acc / l) + x ----
  const float g = gamma[0];
#pragma unroll
  for (int rg = 0; rg < 2; ++rg) {
    v8f linv;
#pragma unroll
    for (int r = 0; r < 8; ++r) linv[r] = 1.0f / sL[rg*16 + r + 8*hi];
#pragma unroll
    for (int cg = 0; cg < 4; ++cg) {
#pragma unroll
      for (int r = 0; r < 8; ++r) {
        long row = qbase + rg*16 + r + 8*hi;
        long off = ((long)bb * NN + row) * CC + w*64 + cg*16 + m;
        out[off] = g * (acc[rg][cg][r] * linv[r]) + x[off];
      }
    }
  }
}

// ---------------- host launcher ----------------
extern "C" void kernel_launch(void* const* d_in, const int* in_sizes, int n_in,
                              void* d_out, int out_size, void* d_ws, size_t ws_size,
                              hipStream_t stream) {
  const float* x  = (const float*)d_in[0];
  const float* Wb = (const float*)d_in[1];
  const float* Wc = (const float*)d_in[2];
  const float* Wd = (const float*)d_in[3];
  const float* gm = (const float*)d_in[4];
  float* out = (float*)d_out;
  char* ws = (char*)d_ws;

  bf16_t* Wt  = (bf16_t*)(ws);                 // 640 KB
  bf16_t* xb  = (bf16_t*)(ws + (1l  << 20));   // 32 MB  bf16 copy of x
  bf16_t* Qp  = (bf16_t*)(ws + (33l << 20));   // 4 MB
  bf16_t* Kp  = (bf16_t*)(ws + (37l << 20));   // 4 MB
  bf16_t* Vtp = (bf16_t*)(ws + (41l << 20));   // 32 MB  V transposed [b][c][n]

  pack_weights<<<(NW * CC + 255) / 256, 256, 0, stream>>>(Wb, Wc, Wd, Wt);
  xcast<<<(NB * NN * CC / 8) / 256, 256, 0, stream>>>(x, xb);
  qkv_proj<<<(NB * NN) / 32, 256, 0, stream>>>(xb, Wt, Qp, Kp, Vtp);
  attn<<<dim3(NN / QT, NB), 256, 57856, stream>>>(Qp, Kp, Vtp, x, gm, out);
}
